// MultiServiceModel_52493090292186
// MI455X (gfx1250) — compile-verified, matrix operand order checked by reading, not run on hardware
//
#include <hip/hip_runtime.h>
#include <hip/hip_bf16.h>

#define T_DIM 8192
#define H_DIM 400
#define IN_DIM 55
#define KCAT 800        // 2H, K of deep-cell GEMM
#define NBIG 2000       // 5 gate chunks * H
#define NOUT 162        // 55 + 52 + 55
#define NOUT_PAD 176    // 11 * 16

typedef __attribute__((ext_vector_type(8)))  _Float16 v8h_t;
typedef __attribute__((ext_vector_type(16))) _Float16 v16h;
typedef __attribute__((ext_vector_type(8)))  float    v8f;

__device__ __forceinline__ float sigf(float x) { return 1.0f / (1.0f + __expf(-x)); }

// Load a 16x32 f16 WMMA fragment (A from row-major [M,K], or B from the
// transposed weight W^T stored row-major [N,K]).
// Per ISA 16-bit A layout: lane half h, VGPR v0..3 <- K = 8h + 0..7,
// VGPR v4..7 <- K = 16 + 8h + 0..7  -> two contiguous 16-byte loads.
__device__ __forceinline__ v16h load_frag16(const _Float16* __restrict__ base,
                                            int ld, int r0, int k0, int lane) {
  const int r  = r0 + (lane & 15);
  const int hv = (lane >> 4) & 1;
  const _Float16* p = base + (size_t)r * ld + (size_t)(k0 + hv * 8);
  v8h_t lo = *(const v8h_t*)p;
  v8h_t hi = *(const v8h_t*)(p + 16);
  v16h f;
#pragma unroll
  for (int i = 0; i < 8; ++i) { f[i] = lo[i]; f[i + 8] = hi[i]; }
  return f;
}

__device__ __forceinline__ v8f wmma32(v16h a, v16h b, v8f c) {
  return __builtin_amdgcn_wmma_f32_16x16x32_f16(false, a, false, b, (short)0, c,
                                                false, false);
}

// ---------------- prep kernels (fp32 -> fp16, transpose/fold) ----------------

__global__ void prep_x16(const float* __restrict__ x, _Float16* __restrict__ y) {
  int idx = blockIdx.x * blockDim.x + threadIdx.x;
  if (idx >= T_DIM * 64) return;
  int t = idx >> 6, k = idx & 63;
  y[idx] = (k < IN_DIM) ? (_Float16)x[t * IN_DIM + k] : (_Float16)0.0f;
}

// WxT: [1600, 64] <- Wx [55, 1600], zero-padded K
__global__ void prep_wx1(const float* __restrict__ Wx, _Float16* __restrict__ y) {
  int idx = blockIdx.x * blockDim.x + threadIdx.x;
  if (idx >= 1600 * 64) return;
  int n = idx >> 6, k = idx & 63;
  y[idx] = (k < IN_DIM) ? (_Float16)Wx[k * 1600 + n] : (_Float16)0.0f;
}

// Folded deep-cell weight, transposed: WT [2000, 800].
// n: chunk = n/400 in {i, f1, f2, o, u}, jj = n%400.
// k: hsel = k/400 (0 -> h1 rows, 1 -> h2 rows), kk = k%400.
// xsel: which child is the cell input x (0 = h1, 1 = h2).
__global__ void prep_wbig(const float* __restrict__ Wx, const float* __restrict__ U,
                          _Float16* __restrict__ y, int xsel) {
  int idx = blockIdx.x * blockDim.x + threadIdx.x;
  if (idx >= NBIG * KCAT) return;
  int n = idx / KCAT, k = idx % KCAT;
  int chunk = n / H_DIM, jj = n % H_DIM;
  int hsel = k / H_DIM, kk = k % H_DIM;
  int gcol = (chunk == 0) ? jj
           : (chunk <= 2) ? (H_DIM + jj)
           : (chunk == 3) ? (2 * H_DIM + jj) : (3 * H_DIM + jj);
  float v = 0.0f;
  if (chunk == 0 || chunk == 3 || chunk == 4) v += U[kk * 1600 + gcol]; // hs @ U{i,o,u}
  if (chunk == 1 && hsel == 0) v += U[kk * 1600 + H_DIM + jj];          // h1 @ Uf
  if (chunk == 2 && hsel == 1) v += U[kk * 1600 + H_DIM + jj];          // h2 @ Uf
  if (hsel == xsel) v += Wx[kk * 1600 + gcol];                           // x @ Wx
  y[idx] = (_Float16)v;
}

// Fused output weight, transposed + zero-padded: WT [176, 800].
__global__ void prep_wout(const float* __restrict__ Wc, const float* __restrict__ Wac,
                          const float* __restrict__ Wap, _Float16* __restrict__ y) {
  int idx = blockIdx.x * blockDim.x + threadIdx.x;
  if (idx >= NOUT_PAD * KCAT) return;
  int n = idx / KCAT, k = idx % KCAT;
  float v = 0.0f;
  if (n < 55)       { v = Wc[k * 55 + n]; }
  else if (n < 107) { if (k < H_DIM)  v = Wac[k * 52 + (n - 55)]; }
  else if (n < NOUT){ if (k >= H_DIM) v = Wap[(k - H_DIM) * 55 + (n - 107)]; }
  y[idx] = (_Float16)v;
}

// ---------------- layer-1 cell: zero children -> c = i*u, h = o*tanh(c) -----

__global__ void lstm_layer1(const _Float16* __restrict__ x16,  // [T, 64]
                            const _Float16* __restrict__ WxT,  // [1600, 64]
                            const float* __restrict__ bias,    // [1600]
                            _Float16* __restrict__ h_out,      // ld KCAT, pre-offset
                            float* __restrict__ c_out)         // [T, 400]
{
  const int lane = threadIdx.x & 31;
  const int wave = (blockIdx.x * blockDim.x + threadIdx.x) >> 5;
  const int mt = wave / (H_DIM / 16), jt = wave % (H_DIM / 16);
  const int m0 = mt * 16, j0 = jt * 16;
  v8f ai = {}, ao = {}, au = {};
#pragma unroll
  for (int k0 = 0; k0 < 64; k0 += 32) {
    v16h a  = load_frag16(x16, 64, m0, k0, lane);
    v16h bi = load_frag16(WxT, 64, j0, k0, lane);
    v16h bo = load_frag16(WxT, 64, 2 * H_DIM + j0, k0, lane);
    v16h bu = load_frag16(WxT, 64, 3 * H_DIM + j0, k0, lane);
    ai = wmma32(a, bi, ai);
    ao = wmma32(a, bo, ao);
    au = wmma32(a, bu, au);
  }
  const int jj = j0 + (lane & 15);
  const int rbase = m0 + ((lane >> 4) & 1) * 8;
  const float bi = bias[jj], bo = bias[2 * H_DIM + jj], bu = bias[3 * H_DIM + jj];
#pragma unroll
  for (int g = 0; g < 8; ++g) {
    const int row = rbase + g;
    float iv = sigf(ai[g] + bi);
    float ov = sigf(ao[g] + bo);
    float uv = tanhf(au[g] + bu);
    float cv = iv * uv;
    float hvv = ov * tanhf(cv);
    c_out[(size_t)row * H_DIM + jj] = cv;
    h_out[(size_t)row * KCAT + jj] = (_Float16)hvv;
  }
}

// ---------------- deep cell: Z = [h1|h2] @ Wbig, fused LSTM epilogue --------

__global__ void lstm_deep(const _Float16* __restrict__ hcat,  // [T, 800]
                          const float* __restrict__ c1,       // [T, 400]
                          const float* __restrict__ c2,       // [T, 400]
                          const _Float16* __restrict__ WT,    // [2000, 800]
                          const float* __restrict__ bias,     // [1600]
                          _Float16* __restrict__ h_out,       // ld KCAT, pre-offset
                          float* __restrict__ c_out,          // [T, 400] or null
                          int store_c)
{
  const int lane = threadIdx.x & 31;
  const int wave = (blockIdx.x * blockDim.x + threadIdx.x) >> 5;
  const int mt = wave / (H_DIM / 16), jt = wave % (H_DIM / 16);
  const int m0 = mt * 16, j0 = jt * 16;
  v8f acc[5] = {{}, {}, {}, {}, {}};   // i, f1, f2, o, u
#pragma unroll 1
  for (int k0 = 0; k0 < KCAT; k0 += 32) {
    __builtin_prefetch(hcat + (size_t)(m0 + (lane & 15)) * KCAT + k0 + 64, 0, 0);
    v16h a = load_frag16(hcat, KCAT, m0, k0, lane);
#pragma unroll
    for (int g = 0; g < 5; ++g) {
      v16h bf = load_frag16(WT, KCAT, g * H_DIM + j0, k0, lane);
      acc[g] = wmma32(a, bf, acc[g]);
    }
  }
  const int jj = j0 + (lane & 15);
  const int rbase = m0 + ((lane >> 4) & 1) * 8;
  const float bi = bias[jj], bf = bias[H_DIM + jj],
              bo = bias[2 * H_DIM + jj], bu = bias[3 * H_DIM + jj];
#pragma unroll
  for (int g = 0; g < 8; ++g) {
    const int row = rbase + g;
    float iv  = sigf(acc[0][g] + bi);
    float f1v = sigf(acc[1][g] + bf);
    float f2v = sigf(acc[2][g] + bf);
    float ov  = sigf(acc[3][g] + bo);
    float uv  = tanhf(acc[4][g] + bu);
    float cv = iv * uv + f1v * c1[(size_t)row * H_DIM + jj]
                       + f2v * c2[(size_t)row * H_DIM + jj];
    float hvv = ov * tanhf(cv);
    h_out[(size_t)row * KCAT + jj] = (_Float16)hvv;
    if (store_c) c_out[(size_t)row * H_DIM + jj] = cv;
  }
}

// ---------------- fused output heads ----------------------------------------

__global__ void out_proj(const _Float16* __restrict__ hcat, // [T, 800]
                         const _Float16* __restrict__ WT,   // [176, 800]
                         const float* __restrict__ b_cur,
                         const float* __restrict__ b_ac,
                         const float* __restrict__ b_ap,
                         float* __restrict__ out)
{
  const int lane = threadIdx.x & 31;
  const int wave = (blockIdx.x * blockDim.x + threadIdx.x) >> 5;
  const int NT = NOUT_PAD / 16;
  const int mt = wave / NT, nt = wave % NT;
  const int m0 = mt * 16, n0 = nt * 16;
  v8f acc = {};
#pragma unroll 1
  for (int k0 = 0; k0 < KCAT; k0 += 32) {
    v16h a  = load_frag16(hcat, KCAT, m0, k0, lane);
    v16h bb = load_frag16(WT, KCAT, n0, k0, lane);
    acc = wmma32(a, bb, acc);
  }
  const int n = n0 + (lane & 15);
  const int rbase = m0 + ((lane >> 4) & 1) * 8;
#pragma unroll
  for (int g = 0; g < 8; ++g) {
    const int row = rbase + g;
    float v = acc[g];
    if (n < 55) {
      out[(size_t)row * 55 + n] = v + b_cur[n];
    } else if (n < 107) {
      out[(size_t)T_DIM * 55 + (size_t)row * 52 + (n - 55)] = v + b_ac[n - 55];
    } else if (n < NOUT) {
      out[(size_t)T_DIM * (55 + 52) + (size_t)row * 55 + (n - 107)] = v + b_ap[n - 107];
    }
  }
}

// ---------------- host --------------------------------------------------------

extern "C" void kernel_launch(void* const* d_in, const int* in_sizes, int n_in,
                              void* d_out, int out_size, void* d_ws, size_t ws_size,
                              hipStream_t stream) {
  (void)in_sizes; (void)n_in; (void)out_size; (void)ws_size;
  const float* x_temp = (const float*)d_in[0];
  const float* x_air  = (const float*)d_in[1];
  const float* Wx_t1 = (const float*)d_in[2];
  const float* U_t1  = (const float*)d_in[3];  (void)U_t1;  // zero children -> unused
  const float* b_t1  = (const float*)d_in[4];
  const float* Wx_t2 = (const float*)d_in[5];
  const float* U_t2  = (const float*)d_in[6];
  const float* b_t2  = (const float*)d_in[7];
  const float* Wx_t3 = (const float*)d_in[8];
  const float* U_t3  = (const float*)d_in[9];
  const float* b_t3  = (const float*)d_in[10];
  const float* Wx_a1 = (const float*)d_in[11];
  const float* U_a1  = (const float*)d_in[12]; (void)U_a1;
  const float* b_a1  = (const float*)d_in[13];
  const float* Wx_a2 = (const float*)d_in[14];
  const float* U_a2  = (const float*)d_in[15];
  const float* b_a2  = (const float*)d_in[16];
  const float* Wx_a3 = (const float*)d_in[17];
  const float* U_a3  = (const float*)d_in[18];
  const float* b_a3  = (const float*)d_in[19];
  const float* W_cur = (const float*)d_in[20];
  const float* b_cur = (const float*)d_in[21];
  const float* W_ac  = (const float*)d_in[22];
  const float* b_ac  = (const float*)d_in[23];
  const float* W_ap  = (const float*)d_in[24];
  const float* b_ap  = (const float*)d_in[25];
  float* out = (float*)d_out;

  char* ws = (char*)d_ws;
  size_t off = 0;
  auto alloc = [&](size_t bytes) -> char* {
    char* p = ws + off;
    off += (bytes + 255) & ~(size_t)255;
    return p;
  };

  _Float16* hcat2 = (_Float16*)alloc((size_t)T_DIM * KCAT * 2);
  _Float16* hcat3 = (_Float16*)alloc((size_t)T_DIM * KCAT * 2);
  _Float16* hcatO = hcat2;  // hcat2 dead after layer 2 -> reuse
  float* cT1 = (float*)alloc((size_t)T_DIM * H_DIM * 4);
  float* cA1 = (float*)alloc((size_t)T_DIM * H_DIM * 4);
  float* cT2 = (float*)alloc((size_t)T_DIM * H_DIM * 4);
  float* cA2 = (float*)alloc((size_t)T_DIM * H_DIM * 4);
  _Float16* x16t = (_Float16*)alloc((size_t)T_DIM * 64 * 2);
  _Float16* x16a = (_Float16*)alloc((size_t)T_DIM * 64 * 2);
  _Float16* wxTt = (_Float16*)alloc((size_t)1600 * 64 * 2);
  _Float16* wxTa = (_Float16*)alloc((size_t)1600 * 64 * 2);
  _Float16* wt2 = (_Float16*)alloc((size_t)NBIG * KCAT * 2);
  _Float16* wa2 = (_Float16*)alloc((size_t)NBIG * KCAT * 2);
  _Float16* wt3 = (_Float16*)alloc((size_t)NBIG * KCAT * 2);
  _Float16* wa3 = (_Float16*)alloc((size_t)NBIG * KCAT * 2);
  _Float16* wout = (_Float16*)alloc((size_t)NOUT_PAD * KCAT * 2);

  // ---- prep
  {
    int n = T_DIM * 64;
    prep_x16<<<(n + 255) / 256, 256, 0, stream>>>(x_temp, x16t);
    prep_x16<<<(n + 255) / 256, 256, 0, stream>>>(x_air,  x16a);
  }
  {
    int n = 1600 * 64;
    prep_wx1<<<(n + 255) / 256, 256, 0, stream>>>(Wx_t1, wxTt);
    prep_wx1<<<(n + 255) / 256, 256, 0, stream>>>(Wx_a1, wxTa);
  }
  {
    int n = NBIG * KCAT;
    prep_wbig<<<(n + 255) / 256, 256, 0, stream>>>(Wx_t2, U_t2, wt2, 0);
    prep_wbig<<<(n + 255) / 256, 256, 0, stream>>>(Wx_a2, U_a2, wa2, 1);
    prep_wbig<<<(n + 255) / 256, 256, 0, stream>>>(Wx_t3, U_t3, wt3, 0);
    prep_wbig<<<(n + 255) / 256, 256, 0, stream>>>(Wx_a3, U_a3, wa3, 1);
  }
  {
    int n = NOUT_PAD * KCAT;
    prep_wout<<<(n + 255) / 256, 256, 0, stream>>>(W_cur, W_ac, W_ap, wout);
  }

  // ---- cells (exact grids: every wave fully active -> EXEC all-ones for WMMA)
  const int waves_cell = (T_DIM / 16) * (H_DIM / 16);   // 12800
  dim3 blk(256);                                        // 8 waves/block
  dim3 grd_cell(waves_cell / 8);                        // 1600

  lstm_layer1<<<grd_cell, blk, 0, stream>>>(x16t, wxTt, b_t1, hcat2 + 0,     cT1);
  lstm_layer1<<<grd_cell, blk, 0, stream>>>(x16a, wxTa, b_a1, hcat2 + H_DIM, cA1);

  lstm_deep<<<grd_cell, blk, 0, stream>>>(hcat2, cT1, cA1, wt2, b_t2,
                                          hcat3 + 0,     cT2, 1);
  lstm_deep<<<grd_cell, blk, 0, stream>>>(hcat2, cT1, cA1, wa2, b_a2,
                                          hcat3 + H_DIM, cA2, 1);

  lstm_deep<<<grd_cell, blk, 0, stream>>>(hcat3, cT2, cA2, wt3, b_t3,
                                          hcatO + 0,     nullptr, 0);
  lstm_deep<<<grd_cell, blk, 0, stream>>>(hcat3, cT2, cA2, wa3, b_a3,
                                          hcatO + H_DIM, nullptr, 0);

  const int waves_out = (T_DIM / 16) * (NOUT_PAD / 16); // 5632
  out_proj<<<waves_out / 8, blk, 0, stream>>>(hcatO, wout, b_cur, b_ac, b_ap, out);
}
